// GCNEncoder_22909355557519
// MI455X (gfx1250) — compile-verified, matrix operand order checked by reading, not run on hardware
//
#include <hip/hip_runtime.h>

typedef __attribute__((ext_vector_type(16))) _Float16 v16h;
typedef __attribute__((ext_vector_type(8)))  float    v8f;

// Problem constants (match reference)
#define N_  50000
#define E_  800000
#define IN_ 32
#define OUT_ 100
#define KP_ 128          // padded K for fp16 fragments (100 -> 128)
#define CT_ 7            // col tiles (7*16 = 112 >= 100)
#define EPS_ 1e-5f

// Workspace layout (byte offsets, all 256-aligned)
#define OFF_STATS 0            // 64 f32: sum[32], sumsq[32]
#define OFF_BI2   1024         // 112 f32 folded bias
#define OFF_WI    2048         // v16h[7][32]  = 7168 B
#define OFF_WH    16384        // v16h[3][4][7][32] = 86016 B
#define OFF_DEG   131072       // N f32
#define OFF_HA    1048576      // N*100 f32
#define OFF_HB    22020096     // N*100 f32
#define OFF_FE    50331648     // E*128 f16 (204.8 MB)

// K index within a 16x32 fp16 A/B fragment for vgpr-pair v (0..7), lane-half hi (0/1)
__device__ __forceinline__ int kbase(int v, int hi) {
    return ((v & 4) << 2) + ((v & 3) << 1) + (hi << 3);
}

__device__ __forceinline__ v8f wmma_f16(v16h a, v16h b, v8f c) {
    return __builtin_amdgcn_wmma_f32_16x16x32_f16(false, a, false, b, (short)0, c, false, false);
}

// ---------------------------------------------------------------- utilities
__global__ void zero_kernel(float* p, long n) {
    long i = (long)blockIdx.x * blockDim.x + threadIdx.x;
    long stride = (long)gridDim.x * blockDim.x;
    for (; i < n; i += stride) p[i] = 0.0f;
}

// ----------------------------------------------------- BatchNorm statistics
__global__ __launch_bounds__(256) void bn_stats_kernel(const float* __restrict__ e,
                                                       float* __restrict__ stats) {
    __shared__ float ls[8 * 64];
    int lane = threadIdx.x & 31;
    int w    = threadIdx.x >> 5;
    int gw   = blockIdx.x * 8 + w;
    int nw   = gridDim.x * 8;
    float s = 0.0f, q = 0.0f;
    for (long r = gw; r < E_; r += nw) {
        float v = e[r * IN_ + lane];   // 32 lanes read one 128B row: coalesced
        s += v; q += v * v;
    }
    ls[w * 64 + lane]      = s;
    ls[w * 64 + 32 + lane] = q;
    __syncthreads();
    if (threadIdx.x < 64) {
        float acc = 0.0f;
        for (int i = 0; i < 8; ++i) acc += ls[i * 64 + threadIdx.x];
        atomicAdd(&stats[threadIdx.x], acc);
    }
}

// ------------------------------------------------------------ degree count
__global__ void deg_count_kernel(const int* __restrict__ dst, float* __restrict__ deg) {
    long i = (long)blockIdx.x * blockDim.x + threadIdx.x;
    long stride = (long)gridDim.x * blockDim.x;
    for (; i < E_; i += stride) atomicAdd(&deg[dst[i]], 1.0f);
}

// -------- fold BN into W_i/bias; pre-pack fp16 B-fragments for all weights
__global__ __launch_bounds__(256) void finalize_kernel(
    const float* __restrict__ stats, const float* __restrict__ gamma,
    const float* __restrict__ beta,  const float* __restrict__ W_i,
    const float* __restrict__ b_i,   const float* __restrict__ W_h,
    float* __restrict__ bi2, v16h* __restrict__ wiFrag, v16h* __restrict__ whFrag) {
    __shared__ float scale[IN_], shift[IN_];
    int t = threadIdx.x;
    if (t < IN_) {
        float mu  = stats[t] * (1.0f / E_);
        float var = stats[32 + t] * (1.0f / E_) - mu * mu;
        float sc  = gamma[t] * rsqrtf(var + EPS_);
        scale[t] = sc;
        shift[t] = beta[t] - mu * sc;
    }
    __syncthreads();
    // folded bias: b_i + shift @ W_i, zero-padded to 112
    for (int c = t; c < 112; c += blockDim.x) {
        float acc = 0.0f;
        if (c < OUT_) {
            acc = b_i[c];
            for (int k = 0; k < IN_; ++k) acc += shift[k] * W_i[k * OUT_ + c];
        }
        bi2[c] = acc;
    }
    // W_i' B-fragments (K=32 exactly, scale folded into rows)
    for (int idx = t; idx < CT_ * 32; idx += blockDim.x) {
        int nt = idx >> 5, l = idx & 31;
        int col = nt * 16 + (l & 15), hi = l >> 4;
        v16h f;
        for (int v = 0; v < 8; ++v) {
            int k = kbase(v, hi);
            float a0 = (col < OUT_) ? scale[k]     * W_i[k * OUT_ + col]       : 0.0f;
            float a1 = (col < OUT_) ? scale[k + 1] * W_i[(k + 1) * OUT_ + col] : 0.0f;
            f[2 * v]     = (_Float16)a0;
            f[2 * v + 1] = (_Float16)a1;
        }
        wiFrag[idx] = f;
    }
    // W_h split into Wa (rows 0..99), Wb (100..199), Wc (201..300); K padded to 128
    for (int idx = t; idx < 3 * 4 * CT_ * 32; idx += blockDim.x) {
        int l = idx & 31;
        int r = idx >> 5;
        int nt = r % CT_; r /= CT_;
        int kt = r & 3;
        int seg = r >> 2;
        int rowbase = (seg == 0) ? 0 : ((seg == 1) ? 100 : 201);
        int col = nt * 16 + (l & 15), hi = l >> 4;
        v16h f;
        for (int v = 0; v < 8; ++v) {
            int k = kt * 32 + kbase(v, hi);
            float a0 = (col < OUT_ && k     < OUT_) ? W_h[(rowbase + k)     * OUT_ + col] : 0.0f;
            float a1 = (col < OUT_ && k + 1 < OUT_) ? W_h[(rowbase + k + 1) * OUT_ + col] : 0.0f;
            f[2 * v]     = (_Float16)a0;
            f[2 * v + 1] = (_Float16)a1;
        }
        whFrag[idx] = f;
    }
}

// -------- f_e = relu(bn(e) @ W_i' + b_i'); store fp16 fragments; scatter f_n
__global__ __launch_bounds__(256) void fe_gemm_kernel(
    const float* __restrict__ e, const int* __restrict__ dst,
    const v16h* __restrict__ wiFrag, const float* __restrict__ bi2,
    _Float16* __restrict__ fe, float* __restrict__ dout) {
    int lane = threadIdx.x & 31, wv = threadIdx.x >> 5;
    long edge0 = (long)blockIdx.x * 16;
    int m = lane & 15, hi = lane >> 4;
    if (wv == 7) {                       // spare wave: zero-pad cols 112..127
        if (lane < 16) {
            uint4 z = {0u, 0u, 0u, 0u};
            uint4* p = (uint4*)(fe + (edge0 + m) * KP_ + 112);
            p[0] = z; p[1] = z;
        }
        return;
    }
    int ct = wv;
    // A fragment: 16 edges x K=32, cvt f32 -> f16 on load
    v16h a;
    const float* erow = e + (edge0 + m) * IN_;
#pragma unroll
    for (int v = 0; v < 8; ++v) {
        int k = kbase(v, hi);
        float2 f = *(const float2*)(erow + k);
        a[2 * v]     = (_Float16)f.x;
        a[2 * v + 1] = (_Float16)f.y;
    }
    v16h b = wiFrag[ct * 32 + lane];
    v8f c = {0.f, 0.f, 0.f, 0.f, 0.f, 0.f, 0.f, 0.f};
    c = wmma_f16(a, b, c);

    int col = ct * 16 + (lane & 15);
    float bias = bi2[col];               // zero for padded cols -> relu(0)=0
#pragma unroll
    for (int j = 0; j < 8; ++j) {
        long er = edge0 + hi * 8 + j;    // C layout: lanes 16-31 hold rows M=8..15
        float vv = c[j] + bias;
        vv = vv > 0.0f ? vv : 0.0f;
        fe[er * KP_ + col] = (_Float16)vv;
        if (col < OUT_) {
            int n = dst[er];
            atomicAdd(&dout[(long)n * 200 + col], vv);   // f_n accumulator = d_out[:, :100]
        }
    }
}

// ---- f_n = accum/deg -> d_out[:, :100] and h0
__global__ void norm_fn_kernel(const float* __restrict__ deg, float* __restrict__ dout,
                               float* __restrict__ h0) {
    long i = (long)blockIdx.x * blockDim.x + threadIdx.x;
    if (i >= (long)N_ * OUT_) return;
    long n = i / OUT_, c = i % OUT_;
    float d = deg[n]; d = d > 1.0f ? d : 1.0f;
    float v = dout[n * 200 + c] / d;
    dout[n * 200 + c] = v;
    h0[i] = v;
}

// -------- one message-passing step: relu(concat(h[src],f_e,p,h[dst]) @ W_h + b_h)
//          -> atomic scatter-sum into h_new
__global__ __launch_bounds__(256) void edge_gemm_kernel(
    const float* __restrict__ hOld, const _Float16* __restrict__ fe,
    const float* __restrict__ p, const float* __restrict__ W_h,
    const float* __restrict__ b_h, const int* __restrict__ src,
    const int* __restrict__ dst, const v16h* __restrict__ whFrag,
    float* __restrict__ hNew) {
    int lane = threadIdx.x & 31, wv = threadIdx.x >> 5;
    if (wv == 7) return;
    int ct = wv;
    long edge0 = (long)blockIdx.x * 16;
    int m = lane & 15, hi = lane >> 4;

    long erow = edge0 + m;
    int nsrc = src[erow], ndst = dst[erow];
    const float* hs = hOld + (long)nsrc * OUT_;
    const float* hd = hOld + (long)ndst * OUT_;
    const unsigned int* feu = (const unsigned int*)(fe + erow * KP_);

    v8f c = {0.f, 0.f, 0.f, 0.f, 0.f, 0.f, 0.f, 0.f};
#pragma unroll
    for (int seg = 0; seg < 3; ++seg) {
        const float* hrow = (seg == 0) ? hs : hd;
#pragma unroll
        for (int kt = 0; kt < 4; ++kt) {
            v16h a;
            if (seg == 1) {
                // f_e already stored as zero-padded fp16: dword pair loads
                union { v16h h; unsigned int u[8]; } ua;
#pragma unroll
                for (int v = 0; v < 8; ++v) {
                    int k = kt * 32 + kbase(v, hi);
                    ua.u[v] = feu[k >> 1];
                }
                a = ua.h;
            } else {
                // gather h row (L2-resident, 20 MB), cvt f32 -> f16
#pragma unroll
                for (int v = 0; v < 8; ++v) {
                    int k = kt * 32 + kbase(v, hi);
                    float2 f;
                    if (k < OUT_) f = *(const float2*)(hrow + k);
                    else { f.x = 0.0f; f.y = 0.0f; }
                    a[2 * v]     = (_Float16)f.x;
                    a[2 * v + 1] = (_Float16)f.y;
                }
            }
            v16h b = whFrag[((seg * 4 + kt) * CT_ + ct) * 32 + lane];
            c = wmma_f16(a, b, c);
        }
    }

    int col = ct * 16 + (lane & 15);
    float bias = (col < OUT_) ? b_h[col] : 0.0f;
    float wpv  = (col < OUT_) ? W_h[200 * OUT_ + col] : 0.0f;   // 'p' row of W_h
#pragma unroll
    for (int j = 0; j < 8; ++j) {
        long er = edge0 + hi * 8 + j;
        float vv = c[j] + bias + p[er] * wpv;
        vv = vv > 0.0f ? vv : 0.0f;
        if (col < OUT_) atomicAdd(&hNew[(long)dst[er] * OUT_ + col], vv);
    }
}

// ---- h = accum/deg (in place); optionally emit final h into d_out[:, 100:200]
__global__ void norm_h_kernel(const float* __restrict__ deg, float* __restrict__ hacc,
                              float* __restrict__ dout, int writeFinal) {
    long i = (long)blockIdx.x * blockDim.x + threadIdx.x;
    if (i >= (long)N_ * OUT_) return;
    long n = i / OUT_, c = i % OUT_;
    float d = deg[n]; d = d > 1.0f ? d : 1.0f;
    float v = hacc[i] / d;
    hacc[i] = v;
    if (writeFinal) dout[n * 200 + 100 + c] = v;
}

// --------------------------------------------------------------- dispatcher
extern "C" void kernel_launch(void* const* d_in, const int* in_sizes, int n_in,
                              void* d_out, int out_size, void* d_ws, size_t ws_size,
                              hipStream_t stream) {
    const float* e     = (const float*)d_in[0];
    const float* p     = (const float*)d_in[1];
    const float* gamma = (const float*)d_in[2];
    const float* beta  = (const float*)d_in[3];
    const float* W_i   = (const float*)d_in[4];
    const float* b_i   = (const float*)d_in[5];
    const float* W_h   = (const float*)d_in[6];
    const float* b_h   = (const float*)d_in[7];
    const int*   src   = (const int*)d_in[8];
    const int*   dst   = (const int*)d_in[9];

    float* dout = (float*)d_out;
    char*  ws   = (char*)d_ws;
    float*     stats  = (float*)(ws + OFF_STATS);
    float*     bi2    = (float*)(ws + OFF_BI2);
    v16h*      wiFrag = (v16h*)(ws + OFF_WI);
    v16h*      whFrag = (v16h*)(ws + OFF_WH);
    float*     deg    = (float*)(ws + OFF_DEG);
    float*     hA     = (float*)(ws + OFF_HA);
    float*     hB     = (float*)(ws + OFF_HB);
    _Float16*  feBuf  = (_Float16*)(ws + OFF_FE);

    const long NH = (long)N_ * OUT_;
    zero_kernel<<<1, 64, 0, stream>>>(stats, 64);
    zero_kernel<<<(N_ + 255) / 256, 256, 0, stream>>>(deg, N_);
    zero_kernel<<<(int)(((long)N_ * 200 + 255) / 256), 256, 0, stream>>>(dout, (long)N_ * 200);

    bn_stats_kernel<<<1024, 256, 0, stream>>>(e, stats);
    deg_count_kernel<<<1024, 256, 0, stream>>>(dst, deg);
    finalize_kernel<<<1, 256, 0, stream>>>(stats, gamma, beta, W_i, b_i, W_h,
                                           bi2, wiFrag, whFrag);

    fe_gemm_kernel<<<E_ / 16, 256, 0, stream>>>(e, dst, wiFrag, bi2, feBuf, dout);
    norm_fn_kernel<<<(int)((NH + 255) / 256), 256, 0, stream>>>(deg, dout, hA);

    float* ho = hA;
    float* hn = hB;
    for (int it = 0; it < 3; ++it) {
        zero_kernel<<<(int)((NH + 255) / 256), 256, 0, stream>>>(hn, NH);
        edge_gemm_kernel<<<E_ / 16, 256, 0, stream>>>(ho, feBuf, p, W_h, b_h, src, dst,
                                                      whFrag, hn);
        norm_h_kernel<<<(int)((NH + 255) / 256), 256, 0, stream>>>(deg, hn, dout,
                                                                   (it == 2) ? 1 : 0);
        float* tmp = ho; ho = hn; hn = tmp;
    }
}
